// Encoder_17824114278582
// MI455X (gfx1250) — matrix operand order repaired and checked by reading, not compiled, not run on hardware
//
#include <hip/hip_runtime.h>

typedef __attribute__((ext_vector_type(16))) _Float16 v16h;
typedef __attribute__((ext_vector_type(8)))  float    v8f;

union AH { v16h h; _Float16 e[16]; unsigned int u[8]; };
union CF { v8f f; float a[8]; };

#define DDIM   128
#define BATCH  8
#define SEQ    2048
#define NROWS  (BATCH*SEQ)

// ---------------------------------------------------------------------------
// Fused: [optional TF-LayerNorm] -> x @ W (128x128) + b -> [optional ReLU]
//        -> [optional + residual] -> [optional += out]
// Block: 128 rows x 128 cols, 256 threads (8 waves), WMMA f16->f32.
// ---------------------------------------------------------------------------
__global__ __launch_bounds__(256) void ln_linear_kernel(
    const float* __restrict__ in, const float* __restrict__ lnw,
    const float* __restrict__ lnb, const float* __restrict__ W,
    const float* __restrict__ bias, const float* __restrict__ res,
    float* __restrict__ out, int ln_en, int relu_en, int accum_en)
{
  __shared__ _Float16 As[128][DDIM];   // normalized input tile, [row][k]
  __shared__ _Float16 Wt[DDIM][DDIM];  // W transposed: Wt[n][k] = W[k][n]

  const int tid  = threadIdx.x;
  const int row0 = blockIdx.x * 128;

  // stage W (d_in-major) transposed into LDS as f16
  for (int i = tid; i < (DDIM*DDIM)/4; i += 256) {
    const int kk = i >> 5;             // d_in row (32 float4 per row)
    const int n4 = (i & 31) << 2;      // d_out col
    const float4 wv = *(const float4*)(W + (size_t)kk*DDIM + n4);
    Wt[n4+0][kk] = (_Float16)wv.x;
    Wt[n4+1][kk] = (_Float16)wv.y;
    Wt[n4+2][kk] = (_Float16)wv.z;
    Wt[n4+3][kk] = (_Float16)wv.w;
  }
  // per-row TF LayerNorm (eps inside sqrt), threads 0..127 own one row each
  if (tid < 128) {
    const float* xr = in + (size_t)(row0 + tid) * DDIM;
    if (ln_en) {
      float s0 = 0.f, s1 = 0.f;
      for (int j = 0; j < DDIM; j += 4) {
        const float4 t4 = *(const float4*)(xr + j);
        s0 += t4.x + t4.y + t4.z + t4.w;
        s1 += t4.x*t4.x + t4.y*t4.y + t4.z*t4.z + t4.w*t4.w;
      }
      const float mean = s0 * (1.0f/DDIM);
      const float var  = s1 * (1.0f/DDIM) - mean*mean;
      const float rstd = rsqrtf(var + 1e-12f);
      for (int j = 0; j < DDIM; j += 4) {
        const float4 t4 = *(const float4*)(xr + j);
        As[tid][j+0] = (_Float16)((t4.x - mean)*rstd*lnw[j+0] + lnb[j+0]);
        As[tid][j+1] = (_Float16)((t4.y - mean)*rstd*lnw[j+1] + lnb[j+1]);
        As[tid][j+2] = (_Float16)((t4.z - mean)*rstd*lnw[j+2] + lnb[j+2]);
        As[tid][j+3] = (_Float16)((t4.w - mean)*rstd*lnw[j+3] + lnb[j+3]);
      }
    } else {
      for (int j = 0; j < DDIM; j += 4) {
        const float4 t4 = *(const float4*)(xr + j);
        As[tid][j+0] = (_Float16)t4.x;
        As[tid][j+1] = (_Float16)t4.y;
        As[tid][j+2] = (_Float16)t4.z;
        As[tid][j+3] = (_Float16)t4.w;
      }
    }
  }
  __syncthreads();

  const int wave = tid >> 5;
  const int lane = tid & 31;
  const int lm   = lane & 15;
  const int lh   = lane >> 4;
  const int mrow = wave * 16;

  CF acc[8];
  for (int n = 0; n < 8; ++n)
    for (int r = 0; r < 8; ++r) acc[n].a[r] = 0.f;

  for (int kk = 0; kk < 4; ++kk) {
    // A fragment: 16-bit A 16x32 layout (lane m = l&15; K split by lane-half)
    AH a;
    const int arow = mrow + lm;
    const int kb0  = kk*32 + lh*16;
    for (int p = 0; p < 4; ++p)
      a.u[p]   = *(const unsigned int*)&As[arow][kb0 + 2*p];
    for (int p = 0; p < 4; ++p)
      a.u[4+p] = *(const unsigned int*)&As[arow][kb0 + 8 + 2*p];
    for (int n = 0; n < 8; ++n) {
      // B fragment: 32x16, lane n = l&15, K = (l>>4)*16 + j
      AH bf;
      const int col = n*16 + lm;
      const int kb  = kk*32 + lh*16;
      for (int p = 0; p < 8; ++p)
        bf.u[p] = *(const unsigned int*)&Wt[col][kb + 2*p];
      acc[n].f = __builtin_amdgcn_wmma_f32_16x16x32_f16(
          false, a.h, false, bf.h, (short)0, acc[n].f, false, false);
    }
  }

  // epilogue: C/D layout row = r + (l>>4)*8, col = n*16 + (l&15)
  for (int n = 0; n < 8; ++n) {
    const int col  = n*16 + lm;
    const float bv = bias[col];
    for (int r = 0; r < 8; ++r) {
      const size_t grow = (size_t)(row0 + mrow + r + lh*8);
      const size_t idx  = grow * DDIM + col;
      float vv = acc[n].a[r] + bv;
      if (relu_en)  vv = fmaxf(vv, 0.f);
      if (res)      vv += res[idx];
      if (accum_en) vv += out[idx];
      out[idx] = vv;
    }
  }
}

// ---------------------------------------------------------------------------
// Flash-style attention with the reference's mask*scale + threshold clip.
// Grid: (SEQ/128, BATCH). Each wave owns 16 query rows; 32-key chunks.
// att = softmax(clip(q@k^T * mask * scale)) @ v + s
// ---------------------------------------------------------------------------
__global__ __launch_bounds__(256) void attn_kernel(
    const float* __restrict__ q, const float* __restrict__ k,
    const float* __restrict__ v, const float* __restrict__ mask,
    const float* __restrict__ sres, float* __restrict__ att)
{
  __shared__ _Float16 Ks[32][DDIM];    // key chunk, [t][d]
  __shared__ _Float16 Vst[DDIM][32];   // value chunk transposed, [d][t]
  __shared__ _Float16 Ps[8][16][32];   // per-wave probs staging (C->A reshape)

  const int tid  = threadIdx.x;
  const int wave = tid >> 5, lane = tid & 31;
  const int lm   = lane & 15, lh = lane >> 4;
  const int b    = blockIdx.y;
  const int q0   = blockIdx.x * 128 + wave * 16;  // q-row base (within batch)
  const float scale = 0.17677669529663687f;       // 1/sqrt(HEAD_SIZE=32)

  // preload q fragments (A layout) for this wave's 16 rows, f32 -> f16
  AH qf[4];
  {
    const float* qr = q + ((size_t)b*SEQ + q0 + lm) * DDIM;
    for (int kq = 0; kq < 4; ++kq)
      for (int p = 0; p < 8; ++p) {
        const int kidx = kq*32 + lh*16 + 2*p;
        qf[kq].e[2*p]   = (_Float16)qr[kidx];
        qf[kq].e[2*p+1] = (_Float16)qr[kidx+1];
      }
  }

  float mrun[8], lsum[8];
  CF oacc[8];
  for (int r = 0; r < 8; ++r) { mrun[r] = -1e30f; lsum[r] = 0.f; }
  for (int n = 0; n < 8; ++n)
    for (int r = 0; r < 8; ++r) oacc[n].a[r] = 0.f;

  for (int t0 = 0; t0 < SEQ; t0 += 32) {
    __syncthreads();
    // cooperative stage: K row-major, V transposed (f16)
    for (int i = tid; i < 32*32; i += 256) {
      const int t  = i >> 5;
      const int d4 = (i & 31) << 2;
      const size_t base = ((size_t)b*SEQ + t0 + t) * DDIM + d4;
      const float4 kv = *(const float4*)(k + base);
      Ks[t][d4+0] = (_Float16)kv.x;  Ks[t][d4+1] = (_Float16)kv.y;
      Ks[t][d4+2] = (_Float16)kv.z;  Ks[t][d4+3] = (_Float16)kv.w;
      const float4 vv = *(const float4*)(v + base);
      Vst[d4+0][t] = (_Float16)vv.x; Vst[d4+1][t] = (_Float16)vv.y;
      Vst[d4+2][t] = (_Float16)vv.z; Vst[d4+3][t] = (_Float16)vv.w;
    }
    __syncthreads();

    // scores: two 16x16 tiles (keys t0..15, t0+16..31), K-depth 128
    CF sc[2];
    for (int tt = 0; tt < 2; ++tt) {
      for (int r = 0; r < 8; ++r) sc[tt].a[r] = 0.f;
      for (int kq = 0; kq < 4; ++kq) {
        AH bf;
        const int key = tt*16 + lm;
        const int kb  = kq*32 + lh*16;
        for (int p = 0; p < 8; ++p)
          bf.u[p] = *(const unsigned int*)&Ks[key][kb + 2*p];
        sc[tt].f = __builtin_amdgcn_wmma_f32_16x16x32_f16(
            false, qf[kq].h, false, bf.h, (short)0, sc[tt].f, false, false);
      }
      // mask * scale, then nonstandard threshold clip
      for (int r = 0; r < 8; ++r) {
        const int qg = q0 + r + lh*8;
        const int tg = t0 + tt*16 + lm;
        const float mv = mask[((size_t)b*SEQ + qg)*SEQ + tg];
        const float x  = sc[tt].a[r] * mv * scale;
        sc[tt].a[r] = (x < 1e-3f) ? -10000.0f : x;
      }
    }

    // online softmax update (row lives in one half-wave's 16 lanes)
    for (int r = 0; r < 8; ++r) {
      float loc = fmaxf(sc[0].a[r], sc[1].a[r]);
      for (int off = 1; off < 16; off <<= 1)
        loc = fmaxf(loc, __shfl_xor(loc, off, 16));
      const float mnew = fmaxf(mrun[r], loc);
      const float sf   = __expf(mrun[r] - mnew);
      const float p0   = __expf(sc[0].a[r] - mnew);
      const float p1   = __expf(sc[1].a[r] - mnew);
      float rs = p0 + p1;
      for (int off = 1; off < 16; off <<= 1)
        rs += __shfl_xor(rs, off, 16);
      lsum[r] = lsum[r]*sf + rs;
      mrun[r] = mnew;
      for (int n = 0; n < 8; ++n) oacc[n].a[r] *= sf;
      Ps[wave][r + lh*8][lm]      = (_Float16)p0;
      Ps[wave][r + lh*8][16 + lm] = (_Float16)p1;
    }

    // reshape probs C-layout -> A-layout via per-wave LDS
    AH pa;
    for (int p = 0; p < 8; ++p) {
      const int kidx = lh*16 + 2*p;
      pa.u[p] = *(const unsigned int*)&Ps[wave][lm][kidx];
    }
    // out += probs(16x32) @ V(32x128)
    for (int n = 0; n < 8; ++n) {
      AH bf;
      const int dcol = n*16 + lm;
      const int kb   = lh*16;
      for (int p = 0; p < 8; ++p)
        bf.u[p] = *(const unsigned int*)&Vst[dcol][kb + 2*p];
      oacc[n].f = __builtin_amdgcn_wmma_f32_16x16x32_f16(
          false, pa.h, false, bf.h, (short)0, oacc[n].f, false, false);
    }
  }

  // finalize: divide by softmax denominator, add residual s
  for (int n = 0; n < 8; ++n) {
    const int col = n*16 + lm;
    for (int r = 0; r < 8; ++r) {
      const size_t grow = (size_t)b*SEQ + q0 + r + lh*8;
      att[grow*DDIM + col] = oacc[n].a[r] / lsum[r] + sres[grow*DDIM + col];
    }
  }
}

// ---------------------------------------------------------------------------
extern "C" void kernel_launch(void* const* d_in, const int* in_sizes, int n_in,
                              void* d_out, int out_size, void* d_ws, size_t ws_size,
                              hipStream_t stream) {
  (void)in_sizes; (void)n_in; (void)out_size; (void)ws_size;
  const float* x         = (const float*)d_in[0];
  const float* mask      = (const float*)d_in[1];
  const float* eop_ln_w  = (const float*)d_in[2];
  const float* eop_ln_b  = (const float*)d_in[3];
  const float* eop_W     = (const float*)d_in[4];
  const float* eop_b     = (const float*)d_in[5];
  const float* attn_ln_w = (const float*)d_in[6];
  const float* attn_ln_b = (const float*)d_in[7];
  const float* Wq = (const float*)d_in[8];
  const float* bq = (const float*)d_in[9];
  const float* Wk = (const float*)d_in[10];
  const float* bk = (const float*)d_in[11];
  const float* Wv = (const float*)d_in[12];
  const float* bv = (const float*)d_in[13];
  const float* ffn_ln_w = (const float*)d_in[14];
  const float* ffn_ln_b = (const float*)d_in[15];
  const float* W1 = (const float*)d_in[16];
  const float* b1 = (const float*)d_in[17];
  const float* W2 = (const float*)d_in[18];
  const float* b2 = (const float*)d_in[19];

  float* ws = (float*)d_ws;
  const size_t NE = (size_t)NROWS * DDIM;
  float* s_   = ws + 0*NE;
  float* q_   = ws + 1*NE;
  float* k_   = ws + 2*NE;
  float* v_   = ws + 3*NE;
  float* att_ = ws + 4*NE;
  float* x1_  = ws + 5*NE;
  float* t_   = q_;  // FFN hidden reuses q buffer (dead after attention)

  const dim3 gG(NROWS/128), bT(256);
  const dim3 gA(SEQ/128, BATCH);

  const float* cur = x;  // ARCH: every edge op reads the previous node output
  for (int i = 0; i < 2; ++i) {
    // s = relu(LN(x)W0+b0) + relu(LN(x)W1+b1) + relu(LN(x)W2+b2)
    for (int oi = 0; oi < 3; ++oi) {
      const int po = i*3 + oi;
      ln_linear_kernel<<<gG, bT, 0, stream>>>(
          cur, eop_ln_w + po*DDIM, eop_ln_b + po*DDIM,
          eop_W + (size_t)po*DDIM*DDIM, eop_b + po*DDIM,
          nullptr, s_, 1, 1, oi > 0);
    }
    // q/k/v = LN(s) @ W + b
    ln_linear_kernel<<<gG, bT, 0, stream>>>(s_, attn_ln_w + i*DDIM, attn_ln_b + i*DDIM,
        Wq + (size_t)i*DDIM*DDIM, bq + i*DDIM, nullptr, q_, 1, 0, 0);
    ln_linear_kernel<<<gG, bT, 0, stream>>>(s_, attn_ln_w + i*DDIM, attn_ln_b + i*DDIM,
        Wk + (size_t)i*DDIM*DDIM, bk + i*DDIM, nullptr, k_, 1, 0, 0);
    ln_linear_kernel<<<gG, bT, 0, stream>>>(s_, attn_ln_w + i*DDIM, attn_ln_b + i*DDIM,
        Wv + (size_t)i*DDIM*DDIM, bv + i*DDIM, nullptr, v_, 1, 0, 0);
    // att = softmax(clip(q k^T * mask * scale)) v + s
    attn_kernel<<<gA, bT, 0, stream>>>(q_, k_, v_, mask, s_, att_);
    // FFN: t = relu(LN(att)W1+b1);  out = t W2 + b2 + att
    ln_linear_kernel<<<gG, bT, 0, stream>>>(att_, ffn_ln_w + i*DDIM, ffn_ln_b + i*DDIM,
        W1 + (size_t)i*DDIM*DDIM, b1 + i*DDIM, nullptr, t_, 1, 1, 0);
    float* outp = (i == 1) ? (float*)d_out : x1_;
    ln_linear_kernel<<<gG, bT, 0, stream>>>(t_, nullptr, nullptr,
        W2 + (size_t)i*DDIM*DDIM, b2 + i*DDIM, att_, outp, 0, 0, 0);
    cur = x1_;
  }
}
